// SARDecoder_18038862643684
// MI455X (gfx1250) — matrix operand order repaired and checked by reading, not compiled, not run on hardware
//
#include <hip/hip_runtime.h>
#include <hip/hip_bf16.h>

// ---------------------------------------------------------------------------
// SAR decoder for MI455X (gfx1250, wave32, WMMA + TDM).
// B=128, H=8, W=32, C=512, RNN=EMB=ATT=512, CLASSES=111, T=31.
// ---------------------------------------------------------------------------

typedef __attribute__((ext_vector_type(16))) _Float16 v16h;
typedef __attribute__((ext_vector_type(8)))  float    v8f;
typedef __attribute__((ext_vector_type(4)))  unsigned u32x4;
typedef __attribute__((ext_vector_type(8)))  int      i32x8;
typedef __attribute__((ext_vector_type(4)))  int      i32x4;

#define Bc   128
#define Hc   8
#define Wc   32
#define Cc   512
#define RNN  512
#define ATT  512
#define CLS  111
#define Tt   31
#define HW   (Hc * Wc)            // 256
#define M_CONV (Bc * HW)          // 32768
#define K_CONV (9 * Cc)           // 4608

#if defined(__gfx1250__) && __has_builtin(__builtin_amdgcn_tensor_load_to_lds)
#define HAVE_TDM 1
#if __has_include(<hip/amd_detail/amd_gfx1250_TDM.h>)
#define TDM_ARITY6 1
#endif
#endif

__device__ __forceinline__ float sigf(float x) { return 1.0f / (1.0f + __expf(-x)); }

__device__ __forceinline__ unsigned rfl(unsigned x) {
  return (unsigned)__builtin_amdgcn_readfirstlane((int)x);
}

// --- WMMA fragment loaders (contiguous k order, same for A and B) -------------------
// A 16x32 f16, row-major source, rowA -> row m at column kk.
// lanes 0-15 (half=0): K=0..15 ; lanes 16-31: K=16..31 (contiguous)
__device__ __forceinline__ v16h load_a_frag(const _Float16* __restrict__ rowA, int half) {
  v16h a;
#pragma unroll
  for (int p = 0; p < 4; ++p) {
    int kb = 16 * half + 2 * p;
    a[2 * p]     = rowA[kb];
    a[2 * p + 1] = rowA[kb + 1];
  }
#pragma unroll
  for (int p = 0; p < 4; ++p) {
    int kb = 16 * half + 8 + 2 * p;
    a[8 + 2 * p]     = rowA[kb];
    a[8 + 2 * p + 1] = rowA[kb + 1];
  }
  return a;
}

// B 32x16 f16 from N-major (transposed) weights: lane n = lane&15,
// K block 16*(lane>>4); 16 contiguous f16 -> two b128 loads.
template <typename PTR>
__device__ __forceinline__ v16h load_b_frag(PTR colB) {
  v16h b;
#pragma unroll
  for (int e = 0; e < 16; ++e) b[e] = colB[e];
  return b;
}

#ifdef HAVE_TDM
// Issue a TDM load of a 2D f16 tile (32 wide x 64 rows, row stride ldb elems)
// from global gptr into LDS byte offset lds_off. D# per CDNA5 ISA ch.8.
__device__ __forceinline__ void tdm_load_b_chunk(const _Float16* gptr, unsigned lds_off,
                                                 int k_rem, int ldb) {
  unsigned long long ga = (unsigned long long)(size_t)gptr;
  u32x4 g0;
  g0[0] = 1u;                                             // count=1, user descriptor
  g0[1] = rfl(lds_off);                                   // lds_addr
  g0[2] = rfl((unsigned)(ga & 0xffffffffu));              // global_addr[31:0]
  g0[3] = rfl((unsigned)((ga >> 32) & 0x01ffffffu)) | (2u << 30);  // addr[56:32]|type=2
  i32x8 g1;
  g1[0] = 0x00010000;                                     // data_size=1 (2 bytes)
  g1[1] = (int)(((unsigned)k_rem & 0xffffu) << 16);       // tensor_dim0 lo16
  g1[2] = (int)((((unsigned)k_rem) >> 16) | (64u << 16)); // dim0 hi16 | tensor_dim1=64
  g1[3] = (int)(32u << 16);                               // dim1 hi=0 | tile_dim0=32
  g1[4] = 64;                                             // tile_dim1=64, tile_dim2=0
  g1[5] = rfl((unsigned)ldb);                             // tensor_dim0_stride lo32
  g1[6] = 0;                                              // stride0 hi | stride1 lo
  g1[7] = 0;
  i32x4 gz = {0, 0, 0, 0};
#ifdef TDM_ARITY6
  i32x8 gz8 = {0, 0, 0, 0, 0, 0, 0, 0};
  __builtin_amdgcn_tensor_load_to_lds(g0, g1, gz, gz, gz8, 0);
#else
  __builtin_amdgcn_tensor_load_to_lds(g0, g1, gz, gz, 0);
#endif
}
#endif // HAVE_TDM

// ---------------------------------------------------------------------------
// Generic WMMA GEMM:  C[M x n_real] = A(f16, lda) * BT(f16, Npad x K)^T + bias
// block = 128 threads (4 waves); block tile 16(M) x 256(N); grid.y = M/16.
// B operand staged through LDS with the Tensor Data Mover, double buffered,
// per-wave (no cross-wave sharing -> TENSORcnt pipelining, no barriers).
// ---------------------------------------------------------------------------
__global__ __launch_bounds__(128) void wmma_gemm(
    const _Float16* __restrict__ A, int lda,
    const _Float16* __restrict__ BT, int ldb,
    const float* __restrict__ bias,
    float* __restrict__ Cf, _Float16* __restrict__ Ch, int ldc,
    int n_real, int K)
{
  const int lane  = threadIdx.x & 31;
  const int wave  = threadIdx.x >> 5;
  const int half  = lane >> 4;
  const int mr    = lane & 15;
  const int m0    = blockIdx.y * 16;
  const int nbase = blockIdx.x * 256 + wave * 64;

  v8f acc[4];
#pragma unroll
  for (int j = 0; j < 4; ++j)
#pragma unroll
    for (int e = 0; e < 8; ++e) acc[j][e] = 0.0f;

  const _Float16* rowA = A + (size_t)(m0 + mr) * lda;

#ifdef HAVE_TDM
  // per-wave double buffer: 64 rows x 32 f16 = 4KB each
  __shared__ _Float16 sB[4][2][64 * 32];
  const _Float16* gB = BT + (size_t)nbase * ldb;
  unsigned lds0 = (unsigned)(size_t)&sB[wave][0][0];
  unsigned lds1 = (unsigned)(size_t)&sB[wave][1][0];
  const int nchunks = K >> 5;

  tdm_load_b_chunk(gB, lds0, K, ldb);
  for (int kc = 0; kc < nchunks; ++kc) {
    const int kk = kc << 5;
    __builtin_prefetch(rowA + kk + 32, 0, 1);            // global_prefetch_b8 (speculative)
    if (kc + 1 < nchunks) {
      tdm_load_b_chunk(gB + (kk + 32), ((kc + 1) & 1) ? lds1 : lds0, K - kk - 32, ldb);
      __builtin_amdgcn_s_wait_tensorcnt(1);              // chunk kc landed, kc+1 in flight
    } else {
      __builtin_amdgcn_s_wait_tensorcnt(0);
    }
    asm volatile("" ::: "memory");                       // LDS written by TDM engine
    const _Float16* sbuf = &sB[wave][kc & 1][0];
    v16h a = load_a_frag(rowA + kk, half);
#pragma unroll
    for (int j = 0; j < 4; ++j) {
      v16h b = load_b_frag(sbuf + (j * 16 + mr) * 32 + 16 * half);  // ds_load_b128 x2
      acc[j] = __builtin_amdgcn_wmma_f32_16x16x32_f16(
          false, a, false, b, (short)0, acc[j], false, false);
    }
  }
#else
  for (int kk = 0; kk < K; kk += 32) {
    __builtin_prefetch(rowA + kk + 32, 0, 1);
    v16h a = load_a_frag(rowA + kk, half);
#pragma unroll
    for (int j = 0; j < 4; ++j) {
      const _Float16* colB = BT + (size_t)(nbase + j * 16 + mr) * ldb + kk + 16 * half;
      v16h b = load_b_frag(colB);
      acc[j] = __builtin_amdgcn_wmma_f32_16x16x32_f16(
          false, a, false, b, (short)0, acc[j], false, false);
    }
  }
#endif

#pragma unroll
  for (int j = 0; j < 4; ++j) {
    int n = nbase + j * 16 + mr;
    if (n < n_real) {
      float bv = bias ? bias[n] : 0.0f;
#pragma unroll
      for (int r = 0; r < 8; ++r) {
        int m = m0 + r + 8 * half;
        float v = acc[j][r] + bv;
        if (Cf) Cf[(size_t)m * ldc + n] = v;
        else    Ch[(size_t)m * ldc + n] = (_Float16)v;
      }
    }
  }
}

// ---------------------------------------------------------------------------
// Implicit-GEMM 3x3 SAME conv: fproj(f16) = conv(features_f16, WfT) + bf.
// M = B*H*W (grid.y = 2048 tiles of 16; 16 rows share b,h since 16|W),
// N = ATT (grid.x = 2), K = 9 taps x 512 channels.
// ---------------------------------------------------------------------------
__global__ __launch_bounds__(128) void conv3x3_wmma(
    const _Float16* __restrict__ Fh,    // (B,H,W,C) f16
    const _Float16* __restrict__ WfT,   // (ATT, 9*C) f16, n-major
    const float* __restrict__ bf,
    _Float16* __restrict__ fproj)       // (B,H,W,ATT) f16
{
  const int lane  = threadIdx.x & 31;
  const int wave  = threadIdx.x >> 5;
  const int half  = lane >> 4;
  const int mr    = lane & 15;
  const int m0    = blockIdx.y * 16;
  const int nbase = blockIdx.x * 256 + wave * 64;

  const int b  = m0 >> 8;
  const int h  = (m0 >> 5) & 7;
  const int w  = (m0 & 31) + mr;       // lane's pixel column

  v8f acc[4];
#pragma unroll
  for (int j = 0; j < 4; ++j)
#pragma unroll
    for (int e = 0; e < 8; ++e) acc[j][e] = 0.0f;

  v16h za;
#pragma unroll
  for (int e = 0; e < 16; ++e) za[e] = (_Float16)0.0f;

  for (int tap = 0; tap < 9; ++tap) {
    const int dy = tap / 3 - 1, dx = tap % 3 - 1;
    const int h2 = h + dy, w2 = w + dx;
    const bool valid = (h2 >= 0) && (h2 < Hc) && (w2 >= 0) && (w2 < Wc);
    const _Float16* rowA = Fh + ((size_t)((b * Hc + h2) * Wc + w2)) * Cc;
    for (int kk = 0; kk < Cc; kk += 32) {
      v16h a = valid ? load_a_frag(rowA + kk, half) : za;
#pragma unroll
      for (int j = 0; j < 4; ++j) {
        const _Float16* colB =
            WfT + (size_t)(nbase + j * 16 + mr) * K_CONV + tap * Cc + kk + 16 * half;
        v16h bb = load_b_frag(colB);
        acc[j] = __builtin_amdgcn_wmma_f32_16x16x32_f16(
            false, a, false, bb, (short)0, acc[j], false, false);
      }
    }
  }

#pragma unroll
  for (int j = 0; j < 4; ++j) {
    int n = nbase + j * 16 + mr;
    float bv = bf[n];
#pragma unroll
    for (int r = 0; r < 8; ++r) {
      int m = m0 + r + 8 * half;
      fproj[(size_t)m * ATT + n] = (_Float16)(acc[j][r] + bv);
    }
  }
}

// --------------------------- setup / elementwise ---------------------------

__global__ void cvt_f16(const float* __restrict__ s, _Float16* __restrict__ d, int n) {
  int i = blockIdx.x * 256 + threadIdx.x;
  if (i < n) d[i] = (_Float16)s[i];
}

__global__ void zero_f16(_Float16* __restrict__ d, int n) {
  int i = blockIdx.x * 256 + threadIdx.x;
  if (i < n) d[i] = (_Float16)0.0f;
}

// src f32 (K x N row-major) -> dst f16 at dst[n*dstStride + colOff + k]
__global__ void transpose_cvt(const float* __restrict__ s, int K, int N,
                              _Float16* __restrict__ d, int dstStride, int colOff) {
  int i = blockIdx.x * 256 + threadIdx.x;
  if (i >= K * N) return;
  int k = i / N, n = i - k * N;
  d[(size_t)n * dstStride + colOff + k] = (_Float16)s[i];
}

__global__ void copy_pad_f32(const float* __restrict__ s, int nsrc,
                             float* __restrict__ d, int ndst) {
  int i = blockIdx.x * 256 + threadIdx.x;
  if (i < ndst) d[i] = (i < nsrc) ? s[i] : 0.0f;
}

// holistic -> cat0 left (f16); zero cat0/cat1 right halves and c0/c1
__global__ void prime_init(const float* __restrict__ hol,
                           _Float16* __restrict__ cat0, _Float16* __restrict__ cat1,
                           float* __restrict__ c0, float* __restrict__ c1) {
  int i = blockIdx.x * 256 + threadIdx.x;
  if (i >= Bc * RNN) return;
  int b = i >> 9, r = i & 511;
  cat0[b * 1024 + r]       = (_Float16)hol[i];
  cat0[b * 1024 + 512 + r] = (_Float16)0.0f;
  cat1[b * 1024 + 512 + r] = (_Float16)0.0f;
  c0[i] = 0.0f;
  c1[i] = 0.0f;
}

// teacher-forced symbol embedding: row gather of embed (one-hot @ W); SOS=111 -> zeros
__global__ void embed_gather(const int* __restrict__ gt, int t,
                             const _Float16* __restrict__ embW,
                             _Float16* __restrict__ cat0) {
  int i = blockIdx.x * 256 + threadIdx.x;
  if (i >= Bc * RNN) return;
  int b = i >> 9, r = i & 511;
  int sym = (t == 0) ? CLS : gt[b * Tt + (t - 1)];
  _Float16 v = (sym >= 0 && sym < CLS) ? embW[sym * 512 + r] : (_Float16)0.0f;
  cat0[b * 1024 + r] = v;
}

// LSTM gate nonlinearity: z (B x 2048) order i,f,g,o -> new c; h written (f16)
// to two cat destinations (row stride 1024) at offsets o1/o2.
__global__ void lstm_ew(const float* __restrict__ z, float* __restrict__ c,
                        _Float16* __restrict__ d1, int o1,
                        _Float16* __restrict__ d2, int o2) {
  int i = blockIdx.x * 256 + threadIdx.x;
  if (i >= Bc * RNN) return;
  int b = i >> 9, r = i & 511;
  const float* zb = z + b * 2048;
  float zi = zb[r], zf = zb[512 + r], zg = zb[1024 + r], zo = zb[1536 + r];
  float c2 = sigf(zf) * c[i] + sigf(zi) * tanhf(zg);
  float h2 = sigf(zo) * tanhf(c2);
  c[i] = c2;
  _Float16 hh = (_Float16)h2;
  d1[b * 1024 + o1 + r] = hh;
  if (d2) d2[b * 1024 + o2 + r] = hh;
}

// attention logits: e[b,p] = sum_c tanh(fproj[b,p,c] + hp[b,c]) * wa[c]
__global__ __launch_bounds__(256) void att_logits(
    const _Float16* __restrict__ fproj, const float* __restrict__ hp,
    const float* __restrict__ wa, float* __restrict__ e) {
  int bp = blockIdx.x;          // b*256 + p
  int b = bp >> 8;
  int tid = threadIdx.x;
  float s = 0.0f;
  for (int c = tid; c < ATT; c += 256)
    s += tanhf((float)fproj[(size_t)bp * ATT + c] + hp[b * ATT + c]) * wa[c];
  __shared__ float red[256];
  red[tid] = s;
  __syncthreads();
  for (int st = 128; st > 0; st >>= 1) {
    if (tid < st) red[tid] += red[tid + st];
    __syncthreads();
  }
  if (tid == 0) e[bp] = red[0];
}

// softmax over 256 positions + glimpse = sum_p att[p]*features; write f16 into
// cat2 right half (logits input).
__global__ __launch_bounds__(256) void att_softmax_glimpse(
    const float* __restrict__ e, const float* __restrict__ features,
    _Float16* __restrict__ cat2) {
  int b = blockIdx.x;
  int tid = threadIdx.x;
  __shared__ float att[256];
  __shared__ float red[256];
  float v = e[b * 256 + tid];
  red[tid] = v;
  __syncthreads();
  for (int st = 128; st > 0; st >>= 1) {
    if (tid < st) red[tid] = fmaxf(red[tid], red[tid + st]);
    __syncthreads();
  }
  float mx = red[0];
  __syncthreads();
  float ex = __expf(v - mx);
  red[tid] = ex;
  __syncthreads();
  for (int st = 128; st > 0; st >>= 1) {
    if (tid < st) red[tid] += red[tid + st];
    __syncthreads();
  }
  float sm = red[0];
  att[tid] = ex / sm;
  __syncthreads();
  for (int c = tid; c < Cc; c += 256) {
    float acc = 0.0f;
    for (int p = 0; p < 256; ++p)
      acc += att[p] * features[((size_t)(b * 256 + p)) * Cc + c];
    cat2[b * 1024 + 512 + c] = (_Float16)acc;
  }
}

// ---------------------------------------------------------------------------
extern "C" void kernel_launch(void* const* d_in, const int* in_sizes, int n_in,
                              void* d_out, int out_size, void* d_ws, size_t ws_size,
                              hipStream_t stream) {
  const float* features = (const float*)d_in[0];
  const float* holistic = (const float*)d_in[1];
  const int*   gt       = (const int*)d_in[2];
  const float* embed_W  = (const float*)d_in[3];
  const float* k0  = (const float*)d_in[4];
  const float* rk0 = (const float*)d_in[5];
  const float* b0  = (const float*)d_in[6];
  const float* k1  = (const float*)d_in[7];
  const float* rk1 = (const float*)d_in[8];
  const float* b1  = (const float*)d_in[9];
  const float* Wh  = (const float*)d_in[10];
  const float* Wf  = (const float*)d_in[11];
  const float* bf  = (const float*)d_in[12];
  const float* wa  = (const float*)d_in[13];
  const float* out_W = (const float*)d_in[14];
  const float* out_b = (const float*)d_in[15];
  float* out = (float*)d_out;

  char* p = (char*)d_ws;
  auto alloc = [&](size_t bytes) -> void* {
    void* r = (void*)p;
    p += (bytes + 255) & ~(size_t)255;
    return r;
  };

  _Float16* Fh   = (_Float16*)alloc((size_t)Bc * HW * Cc * 2);   // features f16
  _Float16* FPJ  = (_Float16*)alloc((size_t)Bc * HW * ATT * 2);  // fproj f16
  _Float16* WfT  = (_Float16*)alloc((size_t)ATT * K_CONV * 2);   // (512,4608)
  _Float16* K0T  = (_Float16*)alloc((size_t)2048 * 1024 * 2);    // [k0;rk0]^T
  _Float16* K1T  = (_Float16*)alloc((size_t)2048 * 1024 * 2);
  _Float16* WhT  = (_Float16*)alloc((size_t)512 * 512 * 2);
  _Float16* OWT  = (_Float16*)alloc((size_t)256 * 1024 * 2);     // out_W^T padded
  _Float16* Eh   = (_Float16*)alloc((size_t)CLS * 512 * 2);
  float*    OBP  = (float*)alloc(256 * 4);                       // out_b padded
  _Float16* cat0 = (_Float16*)alloc((size_t)Bc * 1024 * 2);      // [x  | h0]
  _Float16* cat1 = (_Float16*)alloc((size_t)Bc * 1024 * 2);      // [h0 | h1]
  _Float16* cat2 = (_Float16*)alloc((size_t)Bc * 1024 * 2);      // [h1 | glimpse]
  float*    zbuf = (float*)alloc((size_t)Bc * 2048 * 4);
  float*    c0   = (float*)alloc((size_t)Bc * RNN * 4);
  float*    c1   = (float*)alloc((size_t)Bc * RNN * 4);
  float*    hp   = (float*)alloc((size_t)Bc * ATT * 4);
  float*    ebuf = (float*)alloc((size_t)Bc * 256 * 4);

  auto blks = [](int n) { return (n + 255) / 256; };

  // ---- one-time (per call) setup: f16 conversions + weight transposes ----
  cvt_f16<<<blks(Bc * HW * Cc), 256, 0, stream>>>(features, Fh, Bc * HW * Cc);
  cvt_f16<<<blks(CLS * 512), 256, 0, stream>>>(embed_W, Eh, CLS * 512);
  transpose_cvt<<<blks(K_CONV * ATT), 256, 0, stream>>>(Wf, K_CONV, ATT, WfT, K_CONV, 0);
  transpose_cvt<<<blks(512 * 2048), 256, 0, stream>>>(k0,  512, 2048, K0T, 1024, 0);
  transpose_cvt<<<blks(512 * 2048), 256, 0, stream>>>(rk0, 512, 2048, K0T, 1024, 512);
  transpose_cvt<<<blks(512 * 2048), 256, 0, stream>>>(k1,  512, 2048, K1T, 1024, 0);
  transpose_cvt<<<blks(512 * 2048), 256, 0, stream>>>(rk1, 512, 2048, K1T, 1024, 512);
  transpose_cvt<<<blks(512 * 512), 256, 0, stream>>>(Wh, 512, 512, WhT, 512, 0);
  zero_f16<<<blks(256 * 1024), 256, 0, stream>>>(OWT, 256 * 1024);
  transpose_cvt<<<blks(1024 * CLS), 256, 0, stream>>>(out_W, 1024, CLS, OWT, 1024, 0);
  copy_pad_f32<<<1, 256, 0, stream>>>(out_b, CLS, OBP, 256);

  // ---- conv projector (timestep invariant): WMMA implicit GEMM ----
  conv3x3_wmma<<<dim3(ATT / 256, M_CONV / 16), 128, 0, stream>>>(Fh, WfT, bf, FPJ);

  // ---- prime the stacked LSTM with the holistic vector ----
  prime_init<<<blks(Bc * RNN), 256, 0, stream>>>(holistic, cat0, cat1, c0, c1);
  wmma_gemm<<<dim3(2048 / 256, Bc / 16), 128, 0, stream>>>(
      cat0, 1024, K0T, 1024, b0, zbuf, nullptr, 2048, 2048, 1024);
  lstm_ew<<<blks(Bc * RNN), 256, 0, stream>>>(zbuf, c0, cat1, 0, cat0, 512);
  wmma_gemm<<<dim3(2048 / 256, Bc / 16), 128, 0, stream>>>(
      cat1, 1024, K1T, 1024, b1, zbuf, nullptr, 2048, 2048, 1024);
  lstm_ew<<<blks(Bc * RNN), 256, 0, stream>>>(zbuf, c1, cat1, 512, cat2, 0);

  // ---- teacher-forced decode loop (sequential on stream) ----
  for (int t = 0; t < Tt; ++t) {
    embed_gather<<<blks(Bc * RNN), 256, 0, stream>>>(gt, t, Eh, cat0);

    wmma_gemm<<<dim3(2048 / 256, Bc / 16), 128, 0, stream>>>(
        cat0, 1024, K0T, 1024, b0, zbuf, nullptr, 2048, 2048, 1024);
    lstm_ew<<<blks(Bc * RNN), 256, 0, stream>>>(zbuf, c0, cat1, 0, cat0, 512);

    wmma_gemm<<<dim3(2048 / 256, Bc / 16), 128, 0, stream>>>(
        cat1, 1024, K1T, 1024, b1, zbuf, nullptr, 2048, 2048, 1024);
    lstm_ew<<<blks(Bc * RNN), 256, 0, stream>>>(zbuf, c1, cat1, 512, cat2, 0);

    // hp = h1 @ Wh  (A = cat2 left half, lda 1024)
    wmma_gemm<<<dim3(ATT / 256, Bc / 16), 128, 0, stream>>>(
        cat2, 1024, WhT, 512, nullptr, hp, nullptr, ATT, ATT, 512);

    att_logits<<<Bc * 256, 256, 0, stream>>>(FPJ, hp, wa, ebuf);
    att_softmax_glimpse<<<Bc, 256, 0, stream>>>(ebuf, features, cat2);

    // logits = [h1 | glimpse] @ out_W + out_b -> d_out[:, t, :]
    wmma_gemm<<<dim3(1, Bc / 16), 128, 0, stream>>>(
        cat2, 1024, OWT, 1024, OBP, out + t * CLS, nullptr, Tt * CLS, CLS, 1024);
  }
  (void)in_sizes; (void)n_in; (void)out_size; (void)ws_size;
}
